// MultiDoubleStreamBlockLoraProcessor_52450140619277
// MI455X (gfx1250) — matrix operand order repaired and checked by reading, not compiled, not run on hardware
//
#include <hip/hip_runtime.h>

// ---------------- problem constants ----------------
#define D_MODEL 3072
#define HEADS   24
#define HDIM    128
#define S_TOT   2560     // 512 txt + 2048 hid
#define LTXT    512
#define LHID    2048
#define BLOCKQ  1024     // L_HID - COND
#define IPLEN   64
#define BOUND   1536     // S - COND : mask boundary
#define QK_SCALE 0.08838834764831843f   // 1/sqrt(128)

typedef __attribute__((ext_vector_type(16))) __bf16 v16bf;
typedef __attribute__((ext_vector_type(8)))  __bf16 v8bf;
typedef __attribute__((ext_vector_type(8)))  float  v8f;

static __device__ __forceinline__ unsigned lds_off32(const void* p) {
  // LDS flat aperture: low 32 bits of the generic address are the
  // wave-relative LDS byte offset (ISA 10.2, LDS aperture truncation).
  return (unsigned)(unsigned long long)p;
}

// =====================================================================
// Generic WMMA GEMM (fp32 operands):
//   C[M,N] = alpha * A[M,K] @ B[N,K]^T (+ bias[N])
// fp32 -> bf16 conversion happens while staging into LDS.
// 256 threads = 8 waves; 128x128 tile; K-step 32; wave = 32x64 subtile.
// =====================================================================
__global__ __launch_bounds__(256) void gemm_bf16_wmma(
    const float* __restrict__ A, int lda,
    const float* __restrict__ Bw, int ldb,
    float* __restrict__ C, int ldc,
    const float* __restrict__ bias,
    int M, int N, int K, float alpha)
{
  __shared__ __align__(16) __bf16 As[128][40];  // 32 K + 8 pad (row 80B)
  __shared__ __align__(16) __bf16 Bs[128][40];

  const int tid  = threadIdx.x;
  const int lane = tid & 31;
  const int wave = tid >> 5;
  const int wm   = wave >> 1;   // 0..3
  const int wn   = wave & 1;    // 0..1
  const int bm   = blockIdx.y * 128;
  const int bn   = blockIdx.x * 128;

  v8f acc[2][4] = {};

  const int r    = tid >> 1;
  const int c0   = (tid & 1) * 16;
  const int arow = bm + r;
  const int bcol = bn + r;
  const bool aIn = arow < M;
  const bool bIn = bcol < N;

  for (int kt = 0; kt < K; kt += 32) {
    if (aIn) {
      const float4* s4 = (const float4*)(A + (size_t)arow * lda + kt + c0);
      #pragma unroll
      for (int i = 0; i < 4; ++i) {
        float4 f = s4[i];
        As[r][c0+4*i+0] = (__bf16)f.x; As[r][c0+4*i+1] = (__bf16)f.y;
        As[r][c0+4*i+2] = (__bf16)f.z; As[r][c0+4*i+3] = (__bf16)f.w;
      }
    } else {
      #pragma unroll
      for (int i = 0; i < 16; ++i) As[r][c0+i] = (__bf16)0.f;
    }
    if (bIn) {
      const float4* s4 = (const float4*)(Bw + (size_t)bcol * ldb + kt + c0);
      #pragma unroll
      for (int i = 0; i < 4; ++i) {
        float4 f = s4[i];
        Bs[r][c0+4*i+0] = (__bf16)f.x; Bs[r][c0+4*i+1] = (__bf16)f.y;
        Bs[r][c0+4*i+2] = (__bf16)f.z; Bs[r][c0+4*i+3] = (__bf16)f.w;
      }
    } else {
      #pragma unroll
      for (int i = 0; i < 16; ++i) Bs[r][c0+i] = (__bf16)0.f;
    }
    __syncthreads();

    const int ml = lane & 15;
    const int kh = (lane >> 4) * 8;     // A frag: K halves interleaved
    v16bf afrag[2];
    #pragma unroll
    for (int mi = 0; mi < 2; ++mi) {
      const int m = wm*32 + mi*16 + ml;
      v8bf lo = *(const v8bf*)&As[m][kh];
      v8bf hi = *(const v8bf*)&As[m][16 + kh];
      afrag[mi] = __builtin_shufflevector(lo, hi,
          0,1,2,3,4,5,6,7,8,9,10,11,12,13,14,15);
    }
    const int k0 = (lane >> 4) * 8;     // B frag: lane = column, K halves interleaved
    v16bf bfrag[4];
    #pragma unroll
    for (int ni = 0; ni < 4; ++ni) {
      const int c = wn*64 + ni*16 + ml;
      v8bf lo = *(const v8bf*)&Bs[c][k0];
      v8bf hi = *(const v8bf*)&Bs[c][16 + k0];
      bfrag[ni] = __builtin_shufflevector(lo, hi,
          0,1,2,3,4,5,6,7,8,9,10,11,12,13,14,15);
    }
    #pragma unroll
    for (int mi = 0; mi < 2; ++mi)
      #pragma unroll
      for (int ni = 0; ni < 4; ++ni)
        acc[mi][ni] = __builtin_amdgcn_wmma_f32_16x16x32_bf16(
            false, afrag[mi], false, bfrag[ni],
            (short)0, acc[mi][ni], false, false);
    __syncthreads();
  }

  const int mlc = lane & 15;
  const int mhi = (lane >> 4) * 8;
  #pragma unroll
  for (int mi = 0; mi < 2; ++mi) {
    #pragma unroll
    for (int ni = 0; ni < 4; ++ni) {
      const int nn = bn + wn*64 + ni*16 + mlc;
      if (nn >= N) continue;
      const float bval = bias ? bias[nn] : 0.f;
      const int mbase = bm + wm*32 + mi*16 + mhi;
      #pragma unroll
      for (int j = 0; j < 8; ++j) {
        const int m = mbase + j;
        if (m < M) C[(size_t)m * ldc + nn] = alpha * acc[mi][ni][j] + bval;
      }
    }
  }
}

// =====================================================================
// bf16-operand WMMA GEMM with ASYNC global->LDS staging (attention path).
// All dims are multiples of the tile (M,N mult of 128; K mult of 32):
// no guards -> inner loop is async-copy + ds_load + wmma only.
//   C[M,N] = alpha * A[M,K] @ B[N,K]^T     (A,B bf16; C fp32)
// =====================================================================
__global__ __launch_bounds__(256) void gemm_bf16_async_wmma(
    const __bf16* __restrict__ A, int lda,
    const __bf16* __restrict__ Bw, int ldb,
    float* __restrict__ C, int ldc,
    int M, int N, int K, float alpha)
{
  __shared__ __align__(16) __bf16 As[128][40];  // rows: 64B data + 16B pad
  __shared__ __align__(16) __bf16 Bs[128][40];

  const int tid  = threadIdx.x;
  const int lane = tid & 31;
  const int wave = tid >> 5;
  const int wm   = wave >> 1;
  const int wn   = wave & 1;
  const int bm   = blockIdx.y * 128;
  const int bn   = blockIdx.x * 128;

  v8f acc[2][4] = {};

  for (int kt = 0; kt < K; kt += 32) {
    // --- async stage: 128 rows x 64B = 512 x 16B chunks, 2 per thread ---
    #pragma unroll
    for (int i = 0; i < 2; ++i) {
      const int chunk = tid + 256 * i;
      const int row   = chunk >> 2;
      const int off   = (chunk & 3) * 16;   // byte offset within row
      {
        const unsigned lds = lds_off32(&As[row][0]) + off;
        const unsigned long long g =
            (unsigned long long)(const void*)(A + (size_t)(bm + row) * lda + kt) + off;
        asm volatile("global_load_async_to_lds_b128 %0, %1, off"
                     :: "v"(lds), "v"(g) : "memory");
      }
      {
        const unsigned lds = lds_off32(&Bs[row][0]) + off;
        const unsigned long long g =
            (unsigned long long)(const void*)(Bw + (size_t)(bn + row) * ldb + kt) + off;
        asm volatile("global_load_async_to_lds_b128 %0, %1, off"
                     :: "v"(lds), "v"(g) : "memory");
      }
    }
    asm volatile("s_wait_asynccnt 0x0" ::: "memory");
    __syncthreads();

    const int ml = lane & 15;
    const int kh = (lane >> 4) * 8;
    v16bf afrag[2];
    #pragma unroll
    for (int mi = 0; mi < 2; ++mi) {
      const int m = wm*32 + mi*16 + ml;
      v8bf lo = *(const v8bf*)&As[m][kh];
      v8bf hi = *(const v8bf*)&As[m][16 + kh];
      afrag[mi] = __builtin_shufflevector(lo, hi,
          0,1,2,3,4,5,6,7,8,9,10,11,12,13,14,15);
    }
    const int k0 = (lane >> 4) * 8;
    v16bf bfrag[4];
    #pragma unroll
    for (int ni = 0; ni < 4; ++ni) {
      const int c = wn*64 + ni*16 + ml;
      v8bf lo = *(const v8bf*)&Bs[c][k0];
      v8bf hi = *(const v8bf*)&Bs[c][16 + k0];
      bfrag[ni] = __builtin_shufflevector(lo, hi,
          0,1,2,3,4,5,6,7,8,9,10,11,12,13,14,15);
    }
    #pragma unroll
    for (int mi = 0; mi < 2; ++mi)
      #pragma unroll
      for (int ni = 0; ni < 4; ++ni)
        acc[mi][ni] = __builtin_amdgcn_wmma_f32_16x16x32_bf16(
            false, afrag[mi], false, bfrag[ni],
            (short)0, acc[mi][ni], false, false);
    __syncthreads();
  }

  const int mlc = lane & 15;
  const int mhi = (lane >> 4) * 8;
  #pragma unroll
  for (int mi = 0; mi < 2; ++mi) {
    #pragma unroll
    for (int ni = 0; ni < 4; ++ni) {
      const int nn = bn + wn*64 + ni*16 + mlc;
      const int mbase = bm + wm*32 + mi*16 + mhi;
      #pragma unroll
      for (int j = 0; j < 8; ++j)
        C[(size_t)(mbase + j) * ldc + nn] = alpha * acc[mi][ni][j];
    }
  }
  (void)M;
}

// =====================================================================
// Per-(token,head) RMSNorm in place (fp32). Used for the IP-K path.
// =====================================================================
__global__ __launch_bounds__(128) void rms_rope(
    float* __restrict__ X,
    const float* __restrict__ rc, const float* __restrict__ rs,
    const float* __restrict__ w_lo, const float* __restrict__ w_hi,
    int split, int do_rope, float eps)
{
  const int row = blockIdx.x;
  const int h   = blockIdx.y;
  const int d   = threadIdx.x;
  float* x = X + (size_t)row * D_MODEL + h * HDIM + d;
  float v = *x;

  __shared__ float red[128];
  red[d] = v * v;
  __syncthreads();
  #pragma unroll
  for (int s = 64; s > 0; s >>= 1) {
    if (d < s) red[d] += red[d + s];
    __syncthreads();
  }
  const float r = rsqrtf(red[0] * (1.f / 128.f) + eps);
  float y = v * r;
  const float* w = (row < split) ? w_lo : w_hi;
  if (w) y *= w[d];
  if (do_rope) {
    const float yn  = __shfl_xor(y, 1, 32);
    const float rot = (d & 1) ? yn : -yn;
    y = y * rc[(size_t)row * HDIM + d] + rot * rs[(size_t)row * HDIM + d];
  }
  *x = y;
}

// Same, but reads fp32 and writes bf16 (source left untouched: the IP path
// needs pre-norm Q).  Feeds the async bf16 attention GEMM.
__global__ __launch_bounds__(128) void rms_rope_bf16(
    const float* __restrict__ X, __bf16* __restrict__ Xb,
    const float* __restrict__ rc, const float* __restrict__ rs,
    const float* __restrict__ w_lo, const float* __restrict__ w_hi,
    int split, float eps)
{
  const int row = blockIdx.x;
  const int h   = blockIdx.y;
  const int d   = threadIdx.x;
  const size_t idx = (size_t)row * D_MODEL + h * HDIM + d;
  float v = X[idx];

  __shared__ float red[128];
  red[d] = v * v;
  __syncthreads();
  #pragma unroll
  for (int s = 64; s > 0; s >>= 1) {
    if (d < s) red[d] += red[d + s];
    __syncthreads();
  }
  const float r = rsqrtf(red[0] * (1.f / 128.f) + eps);
  float y = v * r * ((row < split) ? w_lo[d] : w_hi[d]);
  const float yn  = __shfl_xor(y, 1, 32);
  const float rot = (d & 1) ? yn : -yn;
  y = y * rc[(size_t)row * HDIM + d] + rot * rs[(size_t)row * HDIM + d];
  Xb[idx] = (__bf16)y;
}

// =====================================================================
// In-place masked softmax; optional bf16 copy of the probabilities.
// Row i >= boundary only attends to cols >= boundary (boundary<=0: none).
// =====================================================================
__global__ __launch_bounds__(256) void softmax_rows(
    float* __restrict__ Sc, __bf16* __restrict__ Pb,
    int width, int ld, int boundary)
{
  const int i   = blockIdx.x;
  const int tid = threadIdx.x;
  float* row = Sc + (size_t)i * ld;
  __bf16* prow = Pb ? Pb + (size_t)i * ld : nullptr;
  const int kstart = (boundary > 0 && i >= boundary) ? boundary : 0;

  __shared__ float red[256];
  float mx = -3.0e38f;
  for (int j = kstart + tid; j < width; j += 256) mx = fmaxf(mx, row[j]);
  red[tid] = mx; __syncthreads();
  #pragma unroll
  for (int s = 128; s > 0; s >>= 1) {
    if (tid < s) red[tid] = fmaxf(red[tid], red[tid + s]);
    __syncthreads();
  }
  mx = red[0];
  __syncthreads();

  float sum = 0.f;
  for (int j = kstart + tid; j < width; j += 256) {
    const float e = __expf(row[j] - mx);
    row[j] = e; sum += e;
  }
  red[tid] = sum; __syncthreads();
  #pragma unroll
  for (int s = 128; s > 0; s >>= 1) {
    if (tid < s) red[tid] += red[tid + s];
    __syncthreads();
  }
  const float inv = 1.f / red[0];
  for (int j = tid; j < width; j += 256) {
    const float p = (j < kstart) ? 0.f : row[j] * inv;
    row[j] = p;
    if (prow) prow[j] = (__bf16)p;
  }
}

// =====================================================================
// Rank-16 LoRA:  Out[r, :] += up @ (down @ X[r, :])  (ALPHA/R*LORA_W = 1)
// =====================================================================
__global__ __launch_bounds__(256) void lora_add(
    const float* __restrict__ X, const float* __restrict__ down,
    const float* __restrict__ up, float* __restrict__ Out)
{
  const int rrow = blockIdx.x;
  const int tid  = threadIdx.x;
  const float* x = X + (size_t)rrow * D_MODEL;
  float* o = Out + (size_t)rrow * D_MODEL;

  __shared__ float red[256];
  __shared__ float tmp[16];
  for (int j = 0; j < 16; ++j) {
    float p = 0.f;
    for (int k = tid; k < D_MODEL; k += 256) p += x[k] * down[j * D_MODEL + k];
    red[tid] = p; __syncthreads();
    #pragma unroll
    for (int s = 128; s > 0; s >>= 1) {
      if (tid < s) red[tid] += red[tid + s];
      __syncthreads();
    }
    if (tid == 0) tmp[j] = red[0];
    __syncthreads();
  }
  for (int n = tid; n < D_MODEL; n += 256) {
    float a = 0.f;
    #pragma unroll
    for (int j = 0; j < 16; ++j) a += up[n * 16 + j] * tmp[j];
    o[n] += a;
  }
}

// Per-head transposes: dst[d, j] = src[j*stride + d]
__global__ __launch_bounds__(256) void transpose_head(
    const float* __restrict__ src, float* __restrict__ dst, int len, int stride)
{
  const int idx = blockIdx.x * 256 + threadIdx.x;
  if (idx >= HDIM * len) return;
  const int d = idx / len, j = idx % len;
  dst[idx] = src[(size_t)j * stride + d];
}

__global__ __launch_bounds__(256) void transpose_head_bf16(
    const float* __restrict__ src, __bf16* __restrict__ dst, int len, int stride)
{
  const int idx = blockIdx.x * 256 + threadIdx.x;
  if (idx >= HDIM * len) return;
  const int d = idx / len, j = idx % len;
  dst[idx] = (__bf16)src[(size_t)j * stride + d];
}

__global__ __launch_bounds__(256) void add_inplace(
    float* __restrict__ dst, const float* __restrict__ src, int n)
{
  const int i = blockIdx.x * 256 + threadIdx.x;
  if (i < n) dst[i] += src[i];
}

// =====================================================================
extern "C" void kernel_launch(void* const* d_in, const int* in_sizes, int n_in,
                              void* d_out, int out_size, void* d_ws, size_t ws_size,
                              hipStream_t stream)
{
  const float* hidden = (const float*)d_in[0];
  const float* enc    = (const float*)d_in[1];
  const float* img    = (const float*)d_in[2];
  const float* rc     = (const float*)d_in[3];
  const float* rs     = (const float*)d_in[4];
  const float* wq  = (const float*)d_in[5];   const float* bq  = (const float*)d_in[6];
  const float* wk  = (const float*)d_in[7];   const float* bk  = (const float*)d_in[8];
  const float* wv  = (const float*)d_in[9];   const float* bv  = (const float*)d_in[10];
  const float* waq = (const float*)d_in[11];  const float* baq = (const float*)d_in[12];
  const float* wak = (const float*)d_in[13];  const float* bak = (const float*)d_in[14];
  const float* wav = (const float*)d_in[15];  const float* bav = (const float*)d_in[16];
  const float* nqw  = (const float*)d_in[17];
  const float* nkw  = (const float*)d_in[18];
  const float* naqw = (const float*)d_in[19];
  const float* nakw = (const float*)d_in[20];
  const float* wo   = (const float*)d_in[21]; const float* bo   = (const float*)d_in[22];
  const float* wadd = (const float*)d_in[23]; const float* badd = (const float*)d_in[24];
  const float* wkip = (const float*)d_in[25];
  const float* wvip = (const float*)d_in[26];
  const float* qd = (const float*)d_in[27];   const float* qu = (const float*)d_in[28];
  const float* kd = (const float*)d_in[29];   const float* ku = (const float*)d_in[30];
  const float* vd = (const float*)d_in[31];   const float* vu = (const float*)d_in[32];
  // d_in[33]/d_in[34] (p_down/p_up): contribution is sliced away -> dead.

  float* out = (float*)d_out;

  // ---------------- workspace layout ----------------
  float* ws   = (float*)d_ws;
  float* Qat  = ws;                                  // 2560*3072 f32 (pre-norm kept)
  float* Kat  = Qat  + (size_t)S_TOT * D_MODEL;
  float* Vat  = Kat  + (size_t)S_TOT * D_MODEL;
  float* Obuf = Vat  + (size_t)S_TOT * D_MODEL;
  float* Sc   = Obuf + (size_t)S_TOT * D_MODEL;      // 2560*2560 f32
  float* IPK  = Sc   + (size_t)S_TOT * S_TOT;        // 64*3072
  float* IPV  = IPK  + (size_t)IPLEN * D_MODEL;
  float* ipVt = IPV  + (size_t)IPLEN * D_MODEL;      // 128*64
  float* ipSc = ipVt + (size_t)HDIM  * IPLEN;        // 1024*64
  float* ipO  = ipSc + (size_t)BLOCKQ * IPLEN;       // 1024*3072
  __bf16* Qb  = (__bf16*)(ipO + (size_t)BLOCKQ * D_MODEL); // 2560*3072 bf16
  __bf16* Kb  = Qb  + (size_t)S_TOT * D_MODEL;
  __bf16* Vtb = Kb  + (size_t)S_TOT * D_MODEL;       // 128*2560 bf16
  __bf16* Pb  = Vtb + (size_t)HDIM  * S_TOT;         // 2560*2560 bf16

  auto gemm = [&](const float* A, int lda, const float* B, int ldb,
                  float* C, int ldc, const float* bias,
                  int M, int N, int K, float alpha) {
    dim3 g((N + 127) / 128, (M + 127) / 128);
    gemm_bf16_wmma<<<g, 256, 0, stream>>>(A, lda, B, ldb, C, ldc, bias, M, N, K, alpha);
  };
  auto gemm_async = [&](const __bf16* A, int lda, const __bf16* B, int ldb,
                        float* C, int ldc, int M, int N, int K, float alpha) {
    dim3 g(N / 128, M / 128);  // exact multiples on this path
    gemm_bf16_async_wmma<<<g, 256, 0, stream>>>(A, lda, B, ldb, C, ldc, M, N, K, alpha);
  };

  // 1) Projections.  Q/K/V rows: 0..511 encoder, 512..2559 hidden.
  gemm(hidden, D_MODEL, wq,  D_MODEL, Qat + (size_t)LTXT * D_MODEL, D_MODEL, bq,  LHID, D_MODEL, D_MODEL, 1.f);
  gemm(hidden, D_MODEL, wk,  D_MODEL, Kat + (size_t)LTXT * D_MODEL, D_MODEL, bk,  LHID, D_MODEL, D_MODEL, 1.f);
  gemm(hidden, D_MODEL, wv,  D_MODEL, Vat + (size_t)LTXT * D_MODEL, D_MODEL, bv,  LHID, D_MODEL, D_MODEL, 1.f);
  gemm(enc,    D_MODEL, waq, D_MODEL, Qat, D_MODEL, baq, LTXT, D_MODEL, D_MODEL, 1.f);
  gemm(enc,    D_MODEL, wak, D_MODEL, Kat, D_MODEL, bak, LTXT, D_MODEL, D_MODEL, 1.f);
  gemm(enc,    D_MODEL, wav, D_MODEL, Vat, D_MODEL, bav, LTXT, D_MODEL, D_MODEL, 1.f);
  gemm(img,    D_MODEL, wkip, D_MODEL, IPK, D_MODEL, nullptr, IPLEN, D_MODEL, D_MODEL, 1.f);
  gemm(img,    D_MODEL, wvip, D_MODEL, IPV, D_MODEL, nullptr, IPLEN, D_MODEL, D_MODEL, 1.f);

  // 2) LoRA adds on COND rows (hidden 1024..2047 -> Q/K/V rows 1536..2559).
  lora_add<<<BLOCKQ, 256, 0, stream>>>(hidden + (size_t)BLOCKQ * D_MODEL, qd, qu,
                                       Qat + (size_t)BOUND * D_MODEL);
  lora_add<<<BLOCKQ, 256, 0, stream>>>(hidden + (size_t)BLOCKQ * D_MODEL, kd, ku,
                                       Kat + (size_t)BOUND * D_MODEL);
  lora_add<<<BLOCKQ, 256, 0, stream>>>(hidden + (size_t)BLOCKQ * D_MODEL, vd, vu,
                                       Vat + (size_t)BOUND * D_MODEL);

  // 3) IP attention on pre-norm oq = Qat rows 512..1535 (before step 4).
  rms_rope<<<dim3(IPLEN, HEADS), 128, 0, stream>>>(IPK, rc, rs, nullptr, nullptr,
                                                   IPLEN, 0, 1e-5f);
  {
    const int tTot = (HDIM * IPLEN + 255) / 256;
    const float* oq = Qat + (size_t)LTXT * D_MODEL;
    for (int h = 0; h < HEADS; ++h) {
      transpose_head<<<tTot, 256, 0, stream>>>(IPV + h * HDIM, ipVt, IPLEN, D_MODEL);
      gemm(oq + h * HDIM, D_MODEL, IPK + h * HDIM, D_MODEL,
           ipSc, IPLEN, nullptr, BLOCKQ, IPLEN, HDIM, QK_SCALE);
      softmax_rows<<<BLOCKQ, 256, 0, stream>>>(ipSc, nullptr, IPLEN, IPLEN, 0);
      gemm(ipSc, IPLEN, ipVt, IPLEN,
           ipO + h * HDIM, D_MODEL, nullptr, BLOCKQ, HDIM, IPLEN, 1.f);
    }
  }

  // 4) RMS + RoPE -> bf16 Q/K for the async attention GEMM.
  rms_rope_bf16<<<dim3(S_TOT, HEADS), 128, 0, stream>>>(Qat, Qb, rc, rs, naqw, nqw, LTXT, 1e-6f);
  rms_rope_bf16<<<dim3(S_TOT, HEADS), 128, 0, stream>>>(Kat, Kb, rc, rs, nakw, nkw, LTXT, 1e-6f);

  // 5) Main attention per head: async bf16 WMMA GEMMs.
  {
    const int tTot = (HDIM * S_TOT + 255) / 256;
    for (int h = 0; h < HEADS; ++h) {
      gemm_async(Qb + h * HDIM, D_MODEL, Kb + h * HDIM, D_MODEL,
                 Sc, S_TOT, S_TOT, S_TOT, HDIM, QK_SCALE);
      softmax_rows<<<S_TOT, 256, 0, stream>>>(Sc, Pb, S_TOT, S_TOT, BOUND);
      transpose_head_bf16<<<tTot, 256, 0, stream>>>(Vat + h * HDIM, Vtb, S_TOT, D_MODEL);
      gemm_async(Pb, S_TOT, Vtb, S_TOT,
                 Obuf + h * HDIM, D_MODEL, S_TOT, HDIM, S_TOT, 1.f);
    }
  }

  // 6) Output projections straight into d_out (+ IP add on hid half).
  gemm(Obuf, D_MODEL, wadd, D_MODEL, out, D_MODEL, badd, LTXT, D_MODEL, D_MODEL, 1.f);
  gemm(Obuf + (size_t)LTXT * D_MODEL, D_MODEL, wo, D_MODEL,
       out + (size_t)LTXT * D_MODEL, D_MODEL, bo, BLOCKQ, D_MODEL, D_MODEL, 1.f);

  const int nAdd = BLOCKQ * D_MODEL;
  add_inplace<<<(nAdd + 255) / 256, 256, 0, stream>>>(
      out + (size_t)LTXT * D_MODEL, ipO, nAdd);

  (void)in_sizes; (void)n_in; (void)out_size; (void)ws_size;
}
